// LocalAttention_23502061044124
// MI455X (gfx1250) — compile-verified, matrix operand order checked by reading, not run on hardware
//
#include <hip/hip_runtime.h>

#define B_ 4
#define K_ 4096
#define C_ 256
#define H_ 128
#define RADIUS 5

typedef __attribute__((ext_vector_type(16))) __bf16 v16bf;
typedef __attribute__((ext_vector_type(8)))  __bf16 v8bf;
typedef __attribute__((ext_vector_type(8)))  float  v8f;
typedef __attribute__((ext_vector_type(4)))  float  v4f;

// ---------------------------------------------------------------------------
// Kernel 1: per-token MLP score via WMMA (bf16 A/B, f32 accum).
//   s[t] = tanh(x[t,:] @ W1 + b1) @ W2 + b2 ;   exps[t] = exp(s[t])
// Block = 256 threads (8 waves). Each wave owns a 16-token tile and the full
// H=128 hidden width (8 N-tiles of 16).
//
// W1 is staged in LDS (64 KB) PRE-SWIZZLED into WMMA B-fragment order:
//   block index blk = kchunk*16 + nt*2 + half   (kchunk: K/32, nt: N/16, half: K16-half)
//   within a block: lane (= kg*16 + nloc) owns 8 contiguous bf16 (16 bytes)
//     holding B[kc + half*16 + kg*8 + i, nt*16 + nloc], i = 0..7
// so each B fragment is exactly two ds_load_b128.
// ---------------------------------------------------------------------------
__global__ __launch_bounds__(256)
void score_kernel(const float* __restrict__ x, const float* __restrict__ W1,
                  const float* __restrict__ b1, const float* __restrict__ W2,
                  const float* __restrict__ b2, float* __restrict__ exps)
{
    __shared__ __align__(16) __bf16 w1s[C_ * H_];   // 64 KB, fragment-swizzled
    const int tid = threadIdx.x;

    // ---- fill: 4096 16-byte blocks; each thread gathers 8 strided W1 values
    for (int o = tid; o < 4096; o += 256) {
        const int lane_o = o & 31;
        const int blk    = o >> 5;            // kchunk*16 + nt*2 + half
        const int half   = blk & 1;
        const int nt     = (blk >> 1) & 7;
        const int kch    = blk >> 4;
        const int kg     = lane_o >> 4;
        const int nloc   = lane_o & 15;
        const int k0     = kch * 32 + half * 16 + kg * 8;
        const int n      = nt * 16 + nloc;
        v8bf v;
        #pragma unroll
        for (int i = 0; i < 8; ++i) v[i] = (__bf16)W1[(k0 + i) * H_ + n];
        *(v8bf*)&w1s[(size_t)o * 8] = v;      // ds_store_b128
    }
    __syncthreads();

    const int wave = tid >> 5;
    const int lane = tid & 31;
    const int m    = lane & 15;     // A-matrix row owned by this lane
    const int kg   = lane >> 4;     // which K-half of the 32-wide fragment
    const int row0 = blockIdx.x * 128 + wave * 16;

    const float* xrow = x + (size_t)(row0 + m) * C_;

    v8f acc[8] = {};                // 8 N-tiles of 16x16 f32 accumulators

    #pragma unroll
    for (int kc = 0; kc < C_; kc += 32) {
        // A fragment: 16x32 bf16; lane m holds K = kg*8+0..7 and 16+kg*8+0..7
        v16bf a;
        #pragma unroll
        for (int i = 0; i < 8; ++i) {
            a[i]     = (__bf16)xrow[kc + kg * 8 + i];
            a[8 + i] = (__bf16)xrow[kc + 16 + kg * 8 + i];
        }
        const int kch_base = (kc >> 5) * 16;           // kchunk*16
        #pragma unroll
        for (int nt = 0; nt < 8; ++nt) {
            // two ds_load_b128: half0 block and half1 block (+256 elements)
            const __bf16* p0 = &w1s[(size_t)(kch_base + nt * 2) * 256 + lane * 8];
            const v8bf h0 = *(const v8bf*)p0;
            const v8bf h1 = *(const v8bf*)(p0 + 256);
            const v16bf bf = __builtin_shufflevector(h0, h1,
                0, 1, 2, 3, 4, 5, 6, 7, 8, 9, 10, 11, 12, 13, 14, 15);
            acc[nt] = __builtin_amdgcn_wmma_f32_16x16x32_bf16(
                false, a, false, bf, (short)0, acc[nt], false, false);
        }
    }

    // Epilogue: s[m] = sum_n tanh(h[m,n] + b1[n]) * W2[n] + b2
    // C/D layout: VGPR r -> M = r + 8*(lane>=16), N = nt*16 + (lane&15)
    float svec[8];
    #pragma unroll
    for (int r = 0; r < 8; ++r) svec[r] = 0.f;
    #pragma unroll
    for (int nt = 0; nt < 8; ++nt) {
        const int n = nt * 16 + (lane & 15);
        const float b1v = b1[n], w2v = W2[n];
        #pragma unroll
        for (int r = 0; r < 8; ++r)
            svec[r] += tanhf(acc[nt][r] + b1v) * w2v;
    }
    // Reduce across the 16 lanes of each half-wave (wave32 shfl)
    #pragma unroll
    for (int off = 1; off < 16; off <<= 1) {
        #pragma unroll
        for (int r = 0; r < 8; ++r)
            svec[r] += __shfl_xor(svec[r], off, 32);
    }

    const int l = lane & 15;
    if (l < 8) {
        const int mm = ((lane >> 4) << 3) + l;   // lanes 0-7 -> m 0..7, lanes 16-23 -> m 8..15
        const float s = svec[l] + b2[0];
        exps[row0 + mm] = __expf(s);
    }
}

// ---------------------------------------------------------------------------
// Kernel 2a/2b: deterministic two-stage column sum  total[b,c] = sum_k x[b,k,c]
// ---------------------------------------------------------------------------
__global__ __launch_bounds__(256)
void colsum_partial(const float* __restrict__ x, float* __restrict__ partial)
{
    const int b = blockIdx.x >> 4;
    const int chunk = blockIdx.x & 15;
    const int c = threadIdx.x;
    const float* base = x + ((size_t)b * K_ + (size_t)chunk * 256) * C_;
    float s = 0.f;
    for (int k = 0; k < 256; ++k) s += base[(size_t)k * C_ + c];
    partial[(size_t)blockIdx.x * C_ + c] = s;
}

__global__ __launch_bounds__(256)
void colsum_combine(const float* __restrict__ partial, float* __restrict__ total)
{
    const int b = blockIdx.x, c = threadIdx.x;
    float s = 0.f;
    for (int ch = 0; ch < 16; ++ch) s += partial[(size_t)(b * 16 + ch) * C_ + c];
    total[b * C_ + c] = s;
}

// ---------------------------------------------------------------------------
// Kernel 3: one block per (b,i). Writes weighted[b,i,:] (banded correction of
// the precomputed column sum) and the 4096-wide weights row with NT stores.
// ---------------------------------------------------------------------------
__global__ __launch_bounds__(256)
void output_kernel(const float* __restrict__ x, const float* __restrict__ exps,
                   const float* __restrict__ total, float* __restrict__ out)
{
    const int t = blockIdx.x;           // flat (b, i)
    const int b = t >> 12;
    const int i = t & (K_ - 1);
    int jlo = i - RADIUS; if (jlo < 0) jlo = 0;
    int jhi = i + RADIUS; if (jhi > K_ - 1) jhi = K_ - 1;

    __shared__ float inv_sh;
    if (threadIdx.x == 0) {
        float d = (float)K_;            // off-band columns contribute exp(0)=1 each
        for (int j = jlo; j <= jhi; ++j) d += exps[b * K_ + j] - 1.0f;
        inv_sh = 1.0f / d;
    }
    __syncthreads();
    const float inv = inv_sh;

    // weighted[b,i,c] = (total[b,c] + sum_band (e^{s_j}-1) x[b,j,c]) * inv
    const int c = threadIdx.x;
    float val = total[b * C_ + c];
    for (int j = jlo; j <= jhi; ++j)
        val += (exps[b * K_ + j] - 1.0f) * x[((size_t)b * K_ + j) * C_ + c];
    out[(size_t)t * C_ + c] = val * inv;

    // weights row: default 1/denom, band positions exp(s_j)/denom. 268 MB
    // total across the dispatch -> stream with non-temporal 128-bit stores.
    float* wrow = out + (size_t)B_ * K_ * C_ + (size_t)t * K_;
    for (int j0 = threadIdx.x * 4; j0 < K_; j0 += 256 * 4) {
        v4f v = { inv, inv, inv, inv };
        if (j0 + 3 >= jlo && j0 <= jhi) {
            #pragma unroll
            for (int e = 0; e < 4; ++e) {
                const int j = j0 + e;
                if (j >= jlo && j <= jhi) v[e] = exps[b * K_ + j] * inv;
            }
        }
        __builtin_nontemporal_store(v, (v4f*)(wrow + j0));
    }
}

// ---------------------------------------------------------------------------
extern "C" void kernel_launch(void* const* d_in, const int* in_sizes, int n_in,
                              void* d_out, int out_size, void* d_ws, size_t ws_size,
                              hipStream_t stream)
{
    const float* x  = (const float*)d_in[0];
    const float* W1 = (const float*)d_in[1];
    const float* b1 = (const float*)d_in[2];
    const float* W2 = (const float*)d_in[3];
    const float* b2 = (const float*)d_in[4];
    float* out = (float*)d_out;

    // scratch layout (all fully overwritten each launch):
    float* exps    = (float*)d_ws;              // B*K        = 16384 f
    float* partial = exps + B_ * K_;            // B*16*C     = 16384 f
    float* total   = partial + B_ * 16 * C_;    // B*C        =  1024 f

    score_kernel  <<<(B_ * K_) / 128, 256, 0, stream>>>(x, W1, b1, W2, b2, exps);
    colsum_partial<<<B_ * 16,         256, 0, stream>>>(x, partial);
    colsum_combine<<<B_,              256, 0, stream>>>(partial, total);
    output_kernel <<<B_ * K_,         256, 0, stream>>>(x, exps, total, out);
}
